// RNNAttentionModel_44470091383027
// MI455X (gfx1250) — matrix operand order repaired and checked by reading, not compiled, not run on hardware
//
#include <hip/hip_runtime.h>

// ---------------------------------------------------------------------------
// RNN (2-layer tanh) + causal attention + FC head for MI455X (gfx1250).
//  k1: whole RNN in one WGP, weights resident in 320KB LDS, fp32 WMMA 16x16x4
//  k2: Q projection, bf16 WMMA 16x16x32 (f32 accum)
//  k3: flash-attention (online softmax) + fused FC epilogue, bf16 WMMA
// ---------------------------------------------------------------------------

#define Bn 64
#define Tn 1024
#define En 64
#define Hn 128
#define Vo 55

typedef __attribute__((ext_vector_type(2)))  float        v2f;
typedef __attribute__((ext_vector_type(8)))  float        v8f;
typedef __attribute__((ext_vector_type(16))) __bf16       v16bf;
typedef __attribute__((ext_vector_type(8)))  unsigned int v8u;

__device__ __forceinline__ unsigned short f2bfu(float f) {
  unsigned u = __builtin_bit_cast(unsigned, f);
  u += 0x7FFFu + ((u >> 16) & 1u);           // round-to-nearest-even
  return (unsigned short)(u >> 16);
}
__device__ __forceinline__ __bf16 u2bf(unsigned short s) {
  return __builtin_bit_cast(__bf16, s);
}

// branchless tanh: v_exp_f32 + v_rcp_f32, no EXEC-mask games on the RNN
// critical path (clamp keeps exp finite, so no inf/NaN fixup needed)
__device__ __forceinline__ float fast_tanhf(float x) {
  const float xc = fminf(fmaxf(x, -15.f), 15.f);
  const float e  = __expf(2.f * xc);
  return (e - 1.f) * __builtin_amdgcn_rcpf(e + 1.f);
}

// two 16B-contiguous bf16x8 chunks -> one v16bf fragment (2x b128 loads)
__device__ __forceinline__ v16bf ld_bf16x16(const unsigned short* p0,
                                            const unsigned short* p1) {
  const uint4 lo = *(const uint4*)p0;
  const uint4 hi = *(const uint4*)p1;
  v8u u;
  u[0] = lo.x; u[1] = lo.y; u[2] = lo.z; u[3] = lo.w;
  u[4] = hi.x; u[5] = hi.y; u[6] = hi.z; u[7] = hi.w;
  return __builtin_bit_cast(v16bf, u);
}

// ---------------------------------------------------------------------------
// Kernel 1: sequential RNN. 1 block x 1024 threads (32 waves) on one WGP.
// LDS: w0t[192][128], w1t[256][128], bias0/1[128], h0/h1[64][128],
//      emb[55][64], tok[64]  ->  ~303 KB of the 320 KB WGP LDS.
// Each wave owns one 16x16 tile of the [64x128] state GEMM.
// ---------------------------------------------------------------------------
__global__ __launch_bounds__(1024) void rnn_kernel(
    const int* __restrict__ x, const float* __restrict__ emb,
    const float* __restrict__ w_ih0, const float* __restrict__ b_ih0,
    const float* __restrict__ w_hh0, const float* __restrict__ b_hh0,
    const float* __restrict__ w_ih1, const float* __restrict__ b_ih1,
    const float* __restrict__ w_hh1, const float* __restrict__ b_hh1,
    float* __restrict__ outs, unsigned short* __restrict__ outs_bf)
{
  extern __shared__ float smem[];
  float* w0t   = smem;                  // [192*128] w0t[k][n] = concat(w_ih0,w_hh0)[n][k]
  float* w1t   = w0t + 192 * 128;       // [256*128]
  float* bias0 = w1t + 256 * 128;       // [128]
  float* bias1 = bias0 + 128;           // [128]
  float* h0    = bias1 + 128;           // [64*128]
  float* h1    = h0 + 64 * 128;         // [64*128]
  float* embs  = h1 + 64 * 128;         // [55*64]
  int*   tok   = (int*)(embs + 55 * 64);// [64]

  const int tid = threadIdx.x;
  for (int i = tid; i < 192 * 128; i += 1024) {
    int k = i >> 7, n = i & 127;
    w0t[i] = (k < En) ? w_ih0[n * En + k] : w_hh0[n * Hn + (k - En)];
  }
  for (int i = tid; i < 256 * 128; i += 1024) {
    int k = i >> 7, n = i & 127;
    w1t[i] = (k < Hn) ? w_ih1[n * Hn + k] : w_hh1[n * Hn + (k - Hn)];
  }
  if (tid < 128) {
    bias0[tid] = b_ih0[tid] + b_hh0[tid];
    bias1[tid] = b_ih1[tid] + b_hh1[tid];
  }
  for (int i = tid; i < Vo * En; i += 1024) embs[i] = emb[i];
  for (int i = tid; i < 64 * 128; i += 1024) { h0[i] = 0.f; h1[i] = 0.f; }
  __syncthreads();

  const int lane = tid & 31, wv = tid >> 5;
  const int mt = wv >> 3, nt = wv & 7;        // 4 x 8 tile grid of [64x128]
  const int g = lane >> 4, li = lane & 15;
  const int rowA = mt * 16 + li;              // A row = batch index
  const int colB = nt * 16 + li;              // B/C column

  for (int t = 0; t < Tn; ++t) {
    if (tid < Bn) tok[tid] = x[tid * Tn + t];
    __syncthreads();

    // ---- layer 0: h0 = tanh([e | h0] @ [w_ih0|w_hh0]^T + bias0), K = 192
    v8f c;
    const float bsa = bias0[colB];
#pragma unroll
    for (int r = 0; r < 8; ++r) c[r] = bsa;
    const int tkn = tok[rowA];
#pragma unroll 4
    for (int kk = 0; kk < 48; ++kk) {
      const int k0 = kk * 4 + 2 * g;          // f32 A layout: lane half selects k pair
      v2f a, bb;
      if (k0 < En) { a.x = embs[tkn * En + k0];       a.y = embs[tkn * En + k0 + 1]; }
      else         { a.x = h0[rowA * Hn + (k0 - En)]; a.y = h0[rowA * Hn + (k0 - En) + 1]; }
      bb.x = w0t[k0 * 128 + colB];
      bb.y = w0t[(k0 + 1) * 128 + colB];
      c = __builtin_amdgcn_wmma_f32_16x16x4_f32(false, a, false, bb, (short)0, c, false, false);
    }
#pragma unroll
    for (int r = 0; r < 8; ++r) c[r] = fast_tanhf(c[r]);
    __syncthreads();                          // all reads of old h0 done
#pragma unroll
    for (int r = 0; r < 8; ++r) h0[(mt * 16 + r + 8 * g) * Hn + colB] = c[r];
    __syncthreads();

    // ---- layer 1: h1 = tanh([h0_new | h1_old] @ [w_ih1|w_hh1]^T + bias1), K = 256
    v8f c1;
    const float bsb = bias1[colB];
#pragma unroll
    for (int r = 0; r < 8; ++r) c1[r] = bsb;
#pragma unroll 4
    for (int kk = 0; kk < 64; ++kk) {
      const int k0 = kk * 4 + 2 * g;
      v2f a, bb;
      if (k0 < Hn) { a.x = h0[rowA * Hn + k0];        a.y = h0[rowA * Hn + k0 + 1]; }
      else         { a.x = h1[rowA * Hn + (k0 - Hn)]; a.y = h1[rowA * Hn + (k0 - Hn) + 1]; }
      bb.x = w1t[k0 * 128 + colB];
      bb.y = w1t[(k0 + 1) * 128 + colB];
      c1 = __builtin_amdgcn_wmma_f32_16x16x4_f32(false, a, false, bb, (short)0, c1, false, false);
    }
#pragma unroll
    for (int r = 0; r < 8; ++r) c1[r] = fast_tanhf(c1[r]);
    __syncthreads();                          // all reads of old h1 done
#pragma unroll
    for (int r = 0; r < 8; ++r) {
      const int bi = mt * 16 + r + 8 * g;
      const float v = c1[r];
      h1[bi * Hn + colB] = v;
      const size_t off = ((size_t)bi * Tn + t) * Hn + colB;
      outs[off] = v;
      outs_bf[off] = f2bfu(v);
    }
    __syncthreads();
  }
}

// ---------------------------------------------------------------------------
// Kernel 2: Q = outs @ attn_w^T + attn_b   (bf16 WMMA, f32 accumulate)
// one wave per 16-row tile; 4096 blocks x 32 threads
// ---------------------------------------------------------------------------
__global__ __launch_bounds__(32) void qproj_kernel(
    const unsigned short* __restrict__ outs_bf,
    const float* __restrict__ attn_w, const float* __restrict__ attn_b,
    unsigned short* __restrict__ q_bf)
{
  const int lane = threadIdx.x & 31;
  const int g = lane >> 4, li = lane & 15;
  const int bid = blockIdx.x;
  const size_t rowbase = (size_t)(bid >> 6) * Tn + ((bid & 63) << 4);

  // A fragments: 16x32 bf16; lane holds row m=li, two 16B chunks per half
  v16bf afrag[4];
  {
    const unsigned short* arow = outs_bf + (rowbase + li) * Hn;
#pragma unroll
    for (int kc = 0; kc < 4; ++kc)
      afrag[kc] = ld_bf16x16(arow + kc * 32 + 16 * g, arow + kc * 32 + 16 * g + 8);
  }

  for (int ntile = 0; ntile < 8; ++ntile) {
    const int n = ntile * 16 + li;
    v8f c;
    const float bias = attn_b[n];
#pragma unroll
    for (int r = 0; r < 8; ++r) c[r] = bias;
#pragma unroll
    for (int kc = 0; kc < 4; ++kc) {
      v16bf bfrag;                            // B[k][n] = attn_w[n][k], 16 consecutive k
      const float4* wp = (const float4*)(attn_w + n * Hn + kc * 32 + g * 16);
#pragma unroll
      for (int q = 0; q < 4; ++q) {
        const float4 f = wp[q];
        bfrag[q * 4 + 0] = u2bf(f2bfu(f.x));
        bfrag[q * 4 + 1] = u2bf(f2bfu(f.y));
        bfrag[q * 4 + 2] = u2bf(f2bfu(f.z));
        bfrag[q * 4 + 3] = u2bf(f2bfu(f.w));
      }
      c = __builtin_amdgcn_wmma_f32_16x16x32_bf16(false, afrag[kc], false, bfrag,
                                                  (short)0, c, false, false);
    }
#pragma unroll
    for (int r = 0; r < 8; ++r)
      q_bf[(rowbase + r + 8 * g) * Hn + ntile * 16 + li] = f2bfu(c[r]);
  }
}

// ---------------------------------------------------------------------------
// Kernel 3: causal flash attention + fused FC head.
// one wave per (batch b, 16-query tile); KV blocks of 32 staged in LDS.
// ---------------------------------------------------------------------------
__global__ __launch_bounds__(32) void attn_fc_kernel(
    const float* __restrict__ outs,
    const unsigned short* __restrict__ outs_bf,
    const unsigned short* __restrict__ q_bf,
    const float* __restrict__ fc_w, const float* __restrict__ fc_b,
    float* __restrict__ logits)
{
  __shared__ __align__(16) unsigned short kv[32 * Hn];   // 8 KB
  __shared__ __align__(16) unsigned short pbuf[16 * 32]; // 1 KB
  __shared__ __align__(16) float ctx[16 * Hn];           // 8 KB

  const int lane = threadIdx.x & 31;
  const int g = lane >> 4, li = lane & 15;
  const int bid = blockIdx.x;
  const int b = bid >> 6;
  const int t0 = (bid & 63) << 4;
  const size_t rowbase = (size_t)b * Tn + t0;
  const float scale = 0.08838834764831845f;   // 1/sqrt(128)

  v16bf qa[4];
  {
    const unsigned short* qrow = q_bf + (rowbase + li) * Hn;
#pragma unroll
    for (int kc = 0; kc < 4; ++kc)
      qa[kc] = ld_bf16x16(qrow + kc * 32 + 16 * g, qrow + kc * 32 + 16 * g + 8);
  }

  float mrow[8], lrow[8];
  v8f acc[8];
#pragma unroll
  for (int r = 0; r < 8; ++r) { mrow[r] = -3.0e38f; lrow[r] = 0.f; }
#pragma unroll
  for (int hn = 0; hn < 8; ++hn)
#pragma unroll
    for (int r = 0; r < 8; ++r) acc[hn][r] = 0.f;

  for (int s0 = 0; s0 <= t0 + 15; s0 += 32) {
    // stage KV block: lane copies one 256B row
    {
      const uint4* src = (const uint4*)(outs_bf + ((size_t)b * Tn + s0 + lane) * Hn);
      uint4* dst = (uint4*)(kv + lane * Hn);
#pragma unroll
      for (int i = 0; i < 16; ++i) dst[i] = src[i];
    }
    __syncthreads();

    // S[16x32] = Q @ K^T  as two 16x16 tiles
    v8f sc[2];
#pragma unroll
    for (int jh = 0; jh < 2; ++jh) {
#pragma unroll
      for (int r = 0; r < 8; ++r) sc[jh][r] = 0.f;
#pragma unroll
      for (int kc = 0; kc < 4; ++kc) {
        // B[k][j] = outs[s0+jh*16+j][k]: 16 consecutive k per lane
        const unsigned short* kp = kv + (jh * 16 + li) * Hn + kc * 32 + g * 16;
        const v16bf bfrag = ld_bf16x16(kp, kp + 8);
        sc[jh] = __builtin_amdgcn_wmma_f32_16x16x32_bf16(false, qa[kc], false, bfrag,
                                                         (short)0, sc[jh], false, false);
      }
    }

    // online softmax (rows live in the same (r, lane-half) slots for S and acc)
#pragma unroll
    for (int r = 0; r < 8; ++r) {
      const int trow = t0 + r + 8 * g;
      float v0 = sc[0][r] * scale; if (s0 + li > trow)      v0 = -3.0e38f;
      float v1 = sc[1][r] * scale; if (s0 + 16 + li > trow) v1 = -3.0e38f;
      float mt_ = fmaxf(v0, v1);
#pragma unroll
      for (int off = 8; off >= 1; off >>= 1) mt_ = fmaxf(mt_, __shfl_xor(mt_, off, 32));
      const float mnew = fmaxf(mrow[r], mt_);
      const float alpha = __expf(mrow[r] - mnew);
      const float p0 = __expf(v0 - mnew), p1 = __expf(v1 - mnew);
      float rs = p0 + p1;
#pragma unroll
      for (int off = 8; off >= 1; off >>= 1) rs += __shfl_xor(rs, off, 32);
      lrow[r] = lrow[r] * alpha + rs;
      mrow[r] = mnew;
#pragma unroll
      for (int hn = 0; hn < 8; ++hn) acc[hn][r] *= alpha;
      pbuf[(r + 8 * g) * 32 + li]      = f2bfu(p0);
      pbuf[(r + 8 * g) * 32 + 16 + li] = f2bfu(p1);
    }
    __syncthreads();

    // acc += P[16x32] @ V[32x128]
    const unsigned short* pr = pbuf + li * 32 + 16 * g;
    const v16bf pa = ld_bf16x16(pr, pr + 8);
#pragma unroll
    for (int hn = 0; hn < 8; ++hn) {
      v16bf bfrag;                            // B[k][n] = V[s0+k][hn*16+n] (transpose gather)
#pragma unroll
      for (int e = 0; e < 16; ++e)
        bfrag[e] = u2bf(kv[(g * 16 + e) * Hn + hn * 16 + li]);
      acc[hn] = __builtin_amdgcn_wmma_f32_16x16x32_bf16(false, pa, false, bfrag,
                                                        (short)0, acc[hn], false, false);
    }
    __syncthreads();
  }

  // finalize context into LDS
#pragma unroll
  for (int r = 0; r < 8; ++r) {
    const float rl = __builtin_amdgcn_rcpf(lrow[r]);
#pragma unroll
    for (int hn = 0; hn < 8; ++hn)
      ctx[(r + 8 * g) * Hn + hn * 16 + li] = acc[hn][r] * rl;
  }
  __syncthreads();

  // fused FC: logits[i][v] = fc_b[v] + [outs | ctx] . fc_w[v]
  for (int idx = lane; idx < 16 * Vo; idx += 32) {
    const int i = idx / Vo, v = idx % Vo;
    const float4* w4 = (const float4*)(fc_w + v * (2 * Hn));
    const float4* o4 = (const float4*)(outs + (rowbase + i) * Hn);
    const float4* c4 = (const float4*)(ctx + i * Hn);
    float sum = fc_b[v];
#pragma unroll 8
    for (int h4 = 0; h4 < Hn / 4; ++h4) {
      const float4 a = o4[h4], w = w4[h4];
      sum += a.x * w.x + a.y * w.y + a.z * w.z + a.w * w.w;
    }
#pragma unroll 8
    for (int h4 = 0; h4 < Hn / 4; ++h4) {
      const float4 a = c4[h4], w = w4[Hn / 4 + h4];
      sum += a.x * w.x + a.y * w.y + a.z * w.z + a.w * w.w;
    }
    logits[(rowbase + i) * Vo + v] = sum;
  }
}

// ---------------------------------------------------------------------------
extern "C" void kernel_launch(void* const* d_in, const int* in_sizes, int n_in,
                              void* d_out, int out_size, void* d_ws, size_t ws_size,
                              hipStream_t stream) {
  const int*   x      = (const int*)  d_in[0];
  const float* emb    = (const float*)d_in[1];
  const float* w_ih0  = (const float*)d_in[2];
  const float* b_ih0  = (const float*)d_in[3];
  const float* w_hh0  = (const float*)d_in[4];
  const float* b_hh0  = (const float*)d_in[5];
  const float* w_ih1  = (const float*)d_in[6];
  const float* b_ih1  = (const float*)d_in[7];
  const float* w_hh1  = (const float*)d_in[8];
  const float* b_hh1  = (const float*)d_in[9];
  const float* attn_w = (const float*)d_in[10];
  const float* attn_b = (const float*)d_in[11];
  const float* fc_w   = (const float*)d_in[12];
  const float* fc_b   = (const float*)d_in[13];

  const size_t nOuts = (size_t)Bn * Tn * Hn;           // 8,388,608
  float*          outs    = (float*)d_ws;              // 32 MB fp32
  unsigned short* outs_bf = (unsigned short*)(outs + nOuts);  // 16 MB bf16
  unsigned short* q_bf    = outs_bf + nOuts;                  // 16 MB bf16

  const size_t lds1 =
      (192 * 128 + 256 * 128 + 128 + 128 + 64 * 128 + 64 * 128 + 55 * 64) * sizeof(float)
      + 64 * sizeof(int);                              // ~303 KB

  rnn_kernel<<<1, 1024, lds1, stream>>>(x, emb, w_ih0, b_ih0, w_hh0, b_hh0,
                                        w_ih1, b_ih1, w_hh1, b_hh1, outs, outs_bf);

  qproj_kernel<<<Bn * (Tn / 16), 32, 0, stream>>>(outs_bf, attn_w, attn_b, q_bf);

  attn_fc_kernel<<<Bn * (Tn / 16), 32, 0, stream>>>(outs, outs_bf, q_bf,
                                                    fc_w, fc_b, (float*)d_out);
}
